// CustomVimBlock_10196252360762
// MI455X (gfx1250) — matrix operand order repaired and checked by reading, not compiled, not run on hardware
//
#include <hip/hip_runtime.h>
#include <hip/hip_bf16.h>
#include <math.h>

// ---------------- model dims ----------------
#define B_     4
#define CIN    3
#define IMG    384
#define PATCH  16
#define HID    576
#define D_IN   1152
#define NST    16
#define DTR    4
#define KCONV  4
#define MLPD   2304
#define LAYERS 12
#define NCLS   1000
#define GRIDP  24                 // IMG/PATCH
#define LP     (GRIDP*GRIDP)      // 576 patches
#define LL     (LP+1)             // 577 tokens
#define MROWS  (B_*LL)            // 2308
#define MPAIRS ((MROWS+31)/32)    // 73 pairs of 16-row tiles
#define MTILES (MPAIRS*2)         // 146
#define MPAD   (MTILES*16)        // 2336
#define DBLW   64                 // padded width of x-proj output (36 -> 64)
#define CHUNK  32                 // scan chunk length
#define NCH    ((LL+CHUNK-1)/CHUNK) // 19 chunks

typedef _Float16 half_t;
typedef _Float16 v8h  __attribute__((ext_vector_type(8)));
typedef _Float16 v16h __attribute__((ext_vector_type(16)));
typedef float    v8f  __attribute__((ext_vector_type(8)));

// ---------------- device helpers ----------------
__device__ __forceinline__ float silu_f(float x) { return x / (1.f + __expf(-x)); }
__device__ __forceinline__ float gelu_tanh_f(float x) {
    float x3 = x * x * x;
    return 0.5f * x * (1.f + tanhf(0.7978845608028654f * (x + 0.044715f * x3)));
}
__device__ __forceinline__ float softplus_f(float x) {
    return (x > 20.f) ? x : log1pf(__expf(x));
}

// A fragment (16x32 f16): lane holds K = [k0+8*hi, +8) and [k0+16+8*hi, +8) of row m
__device__ __forceinline__ v16h load_frag_a(const half_t* __restrict__ row, int k0, int hi) {
    v8h lo = *(const v8h*)(row + k0 + 8 * hi);
    v8h hh = *(const v8h*)(row + k0 + 16 + 8 * hi);
    return __builtin_shufflevector(lo, hh, 0,1,2,3,4,5,6,7,8,9,10,11,12,13,14,15);
}
// B fragment (32x16 f16): lane n holds K = [k0+8*hi, +8) and [k0+16+8*hi, +8) of column n (Bt row n)
__device__ __forceinline__ v16h load_frag_b(const half_t* __restrict__ row, int k0, int hi) {
    v8h lo = *(const v8h*)(row + k0 + 8 * hi);
    v8h hh = *(const v8h*)(row + k0 + 16 + 8 * hi);
    return __builtin_shufflevector(lo, hh, 0,1,2,3,4,5,6,7,8,9,10,11,12,13,14,15);
}

// ---------------- blocked WMMA GEMM: C[M,N] = A[M,K](f16) * Bt[N,K](f16)^T -------
// Each wave computes a 32(M) x 64(N) block: 2x4 tiles of 16x16. N % 64 == 0.
// mode 0: Cf = acc
// mode 1: Cf = acc + aux (residual, aux may alias Cf)
// mode 2: Ch = (f16) gelu(acc + aux[col])   (bias)
// mode 3: Cf = acc + aux[col]               (bias)
__global__ __launch_bounds__(256)
void wmma_gemm_kernel(const half_t* __restrict__ A, const half_t* __restrict__ Bt,
                      const float* __restrict__ aux, float* __restrict__ Cf,
                      half_t* __restrict__ Ch, int K, int N, int Mpairs, int mode)
{
    const int wid  = threadIdx.x >> 5;        // wave in block (wave32)
    const int lane = threadIdx.x & 31;
    const int mp   = blockIdx.x * 8 + wid;    // 32-row block index
    if (mp >= Mpairs) return;                 // uniform per-wave exit
    const int nq   = blockIdx.y;              // 64-col block index
    const int m16  = lane & 15;
    const int hi   = lane >> 4;

    const half_t* __restrict__ ar0 = A + (size_t)(mp * 32 + m16) * K;
    const half_t* __restrict__ ar1 = A + (size_t)(mp * 32 + 16 + m16) * K;
    const half_t* __restrict__ br0 = Bt + (size_t)(nq * 64 + m16) * K;
    const half_t* __restrict__ br1 = Bt + (size_t)(nq * 64 + 16 + m16) * K;
    const half_t* __restrict__ br2 = Bt + (size_t)(nq * 64 + 32 + m16) * K;
    const half_t* __restrict__ br3 = Bt + (size_t)(nq * 64 + 48 + m16) * K;

    v8f acc[2][4];
#pragma unroll
    for (int mi = 0; mi < 2; ++mi)
#pragma unroll
        for (int ni = 0; ni < 4; ++ni) acc[mi][ni] = (v8f){};

    for (int k0 = 0; k0 < K; k0 += 32) {
        v16h a0 = load_frag_a(ar0, k0, hi);
        v16h a1 = load_frag_a(ar1, k0, hi);
        v16h b0 = load_frag_b(br0, k0, hi);
        v16h b1 = load_frag_b(br1, k0, hi);
        v16h b2 = load_frag_b(br2, k0, hi);
        v16h b3 = load_frag_b(br3, k0, hi);
        acc[0][0] = __builtin_amdgcn_wmma_f32_16x16x32_f16(false, a0, false, b0, (short)0, acc[0][0], false, false);
        acc[0][1] = __builtin_amdgcn_wmma_f32_16x16x32_f16(false, a0, false, b1, (short)0, acc[0][1], false, false);
        acc[0][2] = __builtin_amdgcn_wmma_f32_16x16x32_f16(false, a0, false, b2, (short)0, acc[0][2], false, false);
        acc[0][3] = __builtin_amdgcn_wmma_f32_16x16x32_f16(false, a0, false, b3, (short)0, acc[0][3], false, false);
        acc[1][0] = __builtin_amdgcn_wmma_f32_16x16x32_f16(false, a1, false, b0, (short)0, acc[1][0], false, false);
        acc[1][1] = __builtin_amdgcn_wmma_f32_16x16x32_f16(false, a1, false, b1, (short)0, acc[1][1], false, false);
        acc[1][2] = __builtin_amdgcn_wmma_f32_16x16x32_f16(false, a1, false, b2, (short)0, acc[1][2], false, false);
        acc[1][3] = __builtin_amdgcn_wmma_f32_16x16x32_f16(false, a1, false, b3, (short)0, acc[1][3], false, false);
    }

#pragma unroll
    for (int mi = 0; mi < 2; ++mi) {
#pragma unroll
        for (int ni = 0; ni < 4; ++ni) {
            const int gc = nq * 64 + ni * 16 + m16;
#pragma unroll
            for (int r = 0; r < 8; ++r) {
                const int gr = mp * 32 + mi * 16 + r + 8 * hi;  // C VGPR r -> M = r + 8*half
                const size_t idx = (size_t)gr * N + gc;
                float v = acc[mi][ni][r];
                if (mode == 0)      { Cf[idx] = v; }
                else if (mode == 1) { Cf[idx] = v + aux[idx]; }
                else if (mode == 2) { Ch[idx] = (half_t)gelu_tanh_f(v + aux[gc]); }
                else                { Cf[idx] = v + aux[gc]; }
            }
        }
    }
}

// ---------------- transpose + fp32 -> f16 cast (weights) ----------------
// src: K x N row-major fp32  ->  dst: Npad x K row-major f16 (cols >= N zero-filled)
__global__ void transpose_cast_kernel(const float* __restrict__ src,
                                      half_t* __restrict__ dst, int K, int N, int Npad)
{
    int idx = blockIdx.x * blockDim.x + threadIdx.x;
    if (idx >= K * Npad) return;
    int n = idx / K, k = idx % K;
    dst[idx] = (n < N) ? (half_t)src[(size_t)k * N + n] : (half_t)0.f;
}

// ---------------- patch embedding (conv stride 16) ----------------
__global__ __launch_bounds__(256)
void patch_embed_kernel(const float* __restrict__ x, const float* __restrict__ pw,
                        const float* __restrict__ pb, const float* __restrict__ pos,
                        float* __restrict__ tok)
{
    __shared__ float patch[CIN * PATCH * PATCH];   // 768 floats
    const int bp = blockIdx.x;                     // b * LP + p
    const int b = bp / LP, p = bp % LP;
    const int py = p / GRIDP, px = p % GRIDP;
    for (int idx = threadIdx.x; idx < CIN * PATCH * PATCH; idx += 256) {
        int ci = idx >> 8, rem = idx & 255, i = rem >> 4, j = rem & 15;
        patch[idx] = x[(((size_t)b * CIN + ci) * IMG + py * PATCH + i) * IMG + px * PATCH + j];
    }
    __syncthreads();
    const int c = blockIdx.y * 256 + threadIdx.x;
    if (c >= HID) return;
    const float* __restrict__ w = pw + (size_t)c * (CIN * PATCH * PATCH);
    float s = 0.f;
#pragma unroll 8
    for (int idx = 0; idx < CIN * PATCH * PATCH; ++idx) s += patch[idx] * w[idx];
    tok[((size_t)b * LL + 1 + p) * HID + c] = s + pb[c] + pos[(size_t)p * HID + c];
}

__global__ void cls_init_kernel(const float* __restrict__ ct, float* __restrict__ tok)
{
    const int b = blockIdx.x;
    const int c = blockIdx.y * 256 + threadIdx.x;
    if (c >= HID) return;
    tok[(size_t)b * LL * HID + c] = ct[c];
}

// ---------------- RMSNorm + cast to f16 (one block per row) ----------------
__global__ __launch_bounds__(256)
void rmsnorm_cast_kernel(const float* __restrict__ tok, const float* __restrict__ w,
                         half_t* __restrict__ out)
{
    const int r = blockIdx.x;
    const float* __restrict__ xr = tok + (size_t)r * HID;
    float s = 0.f;
    for (int c = threadIdx.x; c < HID; c += 256) { float v = xr[c]; s += v * v; }
    __shared__ float sd[256];
    sd[threadIdx.x] = s;
    __syncthreads();
    for (int st = 128; st > 0; st >>= 1) {
        if (threadIdx.x < st) sd[threadIdx.x] += sd[threadIdx.x + st];
        __syncthreads();
    }
    const float inv = rsqrtf(sd[0] / (float)HID + 1e-5f);
    for (int c = threadIdx.x; c < HID; c += 256)
        out[(size_t)r * HID + c] = (half_t)(xr[c] * inv * w[c]);
}

// ---------------- causal depthwise conv1d (K=4) + SiLU; f32 + f16 out ----------------
__global__ void conv_silu_kernel(const float* __restrict__ uz, const float* __restrict__ cw,
                                 const float* __restrict__ cb, float* __restrict__ u,
                                 half_t* __restrict__ u16)
{
    const int gid = blockIdx.x * blockDim.x + threadIdx.x;
    if (gid >= B_ * LL * D_IN) return;
    const int d = gid % D_IN;
    const int bl = gid / D_IN;
    const int l = bl % LL, b = bl / LL;
    float s = cb[d];
#pragma unroll
    for (int k = 0; k < KCONV; ++k) {
        int li = l - (KCONV - 1) + k;
        if (li >= 0)
            s += cw[d * KCONV + k] * uz[((size_t)b * LL + li) * (2 * D_IN) + d];
    }
    const float v = silu_f(s);
    u[(size_t)gid]   = v;
    u16[(size_t)gid] = (half_t)v;
}

// ---------------- dt = softplus(dbl[:, :4] @ w_dt + b_dt) ----------------
__global__ void dt_kernel(const float* __restrict__ dbl, const float* __restrict__ wdt,
                          const float* __restrict__ bdt, float* __restrict__ dt)
{
    const int gid = blockIdx.x * blockDim.x + threadIdx.x;
    if (gid >= MROWS * D_IN) return;
    const int d = gid % D_IN;
    const int r = gid / D_IN;
    float s = bdt[d];
#pragma unroll
    for (int j = 0; j < DTR; ++j) s += dbl[(size_t)r * DBLW + j] * wdt[j * D_IN + d];
    dt[(size_t)gid] = softplus_f(s);
}

// ---------------- parallel selective scan: phase 1 (per-chunk local scan) -------------
// For each (b, chunk, d): run the recurrence inside the chunk with h_in = 0, recording
// Q = local end-state and P = product of decay terms, per state n.
__global__ __launch_bounds__(128)
void scan_chunk_kernel(const float* __restrict__ u, const float* __restrict__ dtb,
                       const float* __restrict__ dbl, const float* __restrict__ A_log,
                       float* __restrict__ Pbuf, float* __restrict__ Qbuf)
{
    const int d  = blockIdx.x * 128 + threadIdx.x;   // D_IN = 9*128
    const int ch = blockIdx.y;
    const int b  = blockIdx.z;
    float A[NST], P[NST], h[NST];
#pragma unroll
    for (int n = 0; n < NST; ++n) {
        A[n] = -__expf(A_log[d * NST + n]);
        P[n] = 1.f; h[n] = 0.f;
    }
    const int l0 = ch * CHUNK;
    const int l1 = (l0 + CHUNK < LL) ? l0 + CHUNK : LL;
    for (int l = l0; l < l1; ++l) {
        const size_t r = (size_t)b * LL + l;
        const float dtv = dtb[r * D_IN + d];
        const float uv  = u[r * D_IN + d];
        const float dtu = dtv * uv;
        const float* __restrict__ dr = dbl + r * DBLW;
#pragma unroll
        for (int n = 0; n < NST; ++n) {
            const float dA = __expf(dtv * A[n]);
            h[n] = dA * h[n] + dtu * dr[DTR + n];
            P[n] *= dA;
        }
    }
    const size_t base = (((size_t)b * NCH + ch) * D_IN + d) * NST;
#pragma unroll
    for (int n = 0; n < NST; ++n) { Pbuf[base + n] = P[n]; Qbuf[base + n] = h[n]; }
}

// ---------------- phase 2: sequential combine over chunk summaries --------------------
// For each (b, d, n): h_in(ch) via h <- P*h + Q across the NCH chunks.
__global__ void scan_combine_kernel(const float* __restrict__ Pbuf,
                                    const float* __restrict__ Qbuf,
                                    float* __restrict__ Hin)
{
    const int gid = blockIdx.x * blockDim.x + threadIdx.x;
    if (gid >= B_ * D_IN * NST) return;
    const int n = gid & (NST - 1);
    const int d = (gid >> 4) % D_IN;
    const int b = gid / (NST * D_IN);
    float h = 0.f;
    for (int ch = 0; ch < NCH; ++ch) {
        const size_t base = (((size_t)b * NCH + ch) * D_IN + d) * NST + n;
        Hin[base] = h;
        h = Pbuf[base] * h + Qbuf[base];
    }
}

// ---------------- phase 3: replay chunk with true h_in, emit gated y (f16) ------------
__global__ __launch_bounds__(128)
void scan_final_kernel(const float* __restrict__ u, const float* __restrict__ dtb,
                       const float* __restrict__ dbl, const float* __restrict__ uz,
                       const float* __restrict__ A_log, const float* __restrict__ Dp,
                       const float* __restrict__ Hin, half_t* __restrict__ yh)
{
    const int d  = blockIdx.x * 128 + threadIdx.x;
    const int ch = blockIdx.y;
    const int b  = blockIdx.z;
    const size_t base = (((size_t)b * NCH + ch) * D_IN + d) * NST;
    float A[NST], h[NST];
#pragma unroll
    for (int n = 0; n < NST; ++n) {
        A[n] = -__expf(A_log[d * NST + n]);
        h[n] = Hin[base + n];
    }
    const float Dv = Dp[d];
    const int l0 = ch * CHUNK;
    const int l1 = (l0 + CHUNK < LL) ? l0 + CHUNK : LL;
    for (int l = l0; l < l1; ++l) {
        const size_t r = (size_t)b * LL + l;
        const float dtv = dtb[r * D_IN + d];
        const float uv  = u[r * D_IN + d];
        const float dtu = dtv * uv;
        const float* __restrict__ dr = dbl + r * DBLW;
        float acc = 0.f;
#pragma unroll
        for (int n = 0; n < NST; ++n) {
            const float dA = __expf(dtv * A[n]);
            h[n] = dA * h[n] + dtu * dr[DTR + n];       // B_m
            acc += h[n] * dr[DTR + NST + n];            // C_m
        }
        float y = acc + uv * Dv;
        const float z = uz[r * (2 * D_IN) + D_IN + d];
        y *= silu_f(z);
        yh[r * D_IN + d] = (half_t)y;
    }
}

// ---------------- gather token 0 of each batch into a 32-row f16 tile pair ------------
__global__ void gather_cls_kernel(const float* __restrict__ tok, half_t* __restrict__ clsin)
{
    const int idx = blockIdx.x * blockDim.x + threadIdx.x;
    if (idx >= 32 * HID) return;
    const int c = idx % HID, r = idx / HID;
    clsin[idx] = (r < B_) ? (half_t)tok[((size_t)r * LL) * HID + c] : (half_t)0.f;
}

// ---------------- classifier head ----------------
__global__ void head_kernel(const float* __restrict__ h2, const float* __restrict__ cw,
                            const float* __restrict__ cb, float* __restrict__ out)
{
    const int gid = blockIdx.x * blockDim.x + threadIdx.x;
    if (gid >= B_ * NCLS) return;
    const int c = gid % NCLS, b = gid / NCLS;
    const float* __restrict__ hr = h2 + (size_t)b * HID;
    float s = cb[c];
    for (int d = 0; d < HID; ++d) s += hr[d] * cw[(size_t)d * NCLS + c];
    out[gid] = s;
}

// ---------------- host launch ----------------
extern "C" void kernel_launch(void* const* d_in, const int* in_sizes, int n_in,
                              void* d_out, int out_size, void* d_ws, size_t ws_size,
                              hipStream_t stream)
{
    (void)in_sizes; (void)n_in; (void)out_size; (void)ws_size;
    const float* x        = (const float*)d_in[0];
    const float* patch_w  = (const float*)d_in[1];
    const float* patch_b  = (const float*)d_in[2];
    const float* pos_emb  = (const float*)d_in[3];
    const float* cls_tok  = (const float*)d_in[4];
    const float* norm_w   = (const float*)d_in[5];
    const float* w_in     = (const float*)d_in[6];
    const float* conv_w   = (const float*)d_in[7];
    const float* conv_b   = (const float*)d_in[8];
    const float* w_xproj  = (const float*)d_in[9];
    const float* w_dt     = (const float*)d_in[10];
    const float* b_dt     = (const float*)d_in[11];
    const float* A_log    = (const float*)d_in[12];
    const float* D_param  = (const float*)d_in[13];
    const float* w_out    = (const float*)d_in[14];
    const float* mlp_w1   = (const float*)d_in[15];
    const float* mlp_b1   = (const float*)d_in[16];
    const float* mlp_w2   = (const float*)d_in[17];
    const float* mlp_b2   = (const float*)d_in[18];
    const float* cls_w    = (const float*)d_in[19];
    const float* cls_b    = (const float*)d_in[20];

    char* ws = (char*)d_ws;
    size_t off = 0;
    auto alloc = [&](size_t bytes) -> void* {
        void* p = ws + off;
        off = (off + bytes + 255) & ~(size_t)255;
        return p;
    };

    const size_t sumN = (size_t)B_ * NCH * D_IN * NST;
    float*  tok    = (float*) alloc((size_t)MPAD * HID * 4);
    half_t* h16    = (half_t*)alloc((size_t)MPAD * HID * 2);
    float*  uz     = (float*) alloc((size_t)MPAD * 2 * D_IN * 4);
    float*  u      = (float*) alloc((size_t)MPAD * D_IN * 4);
    half_t* u16    = (half_t*)alloc((size_t)MPAD * D_IN * 2);
    float*  dbl    = (float*) alloc((size_t)MPAD * DBLW * 4);
    float*  dtb    = (float*) alloc((size_t)MROWS * D_IN * 4);
    half_t* y16    = (half_t*)alloc((size_t)MPAD * D_IN * 2);
    float*  Pbuf   = (float*) alloc(sumN * 4);
    float*  Qbuf   = (float*) alloc(sumN * 4);
    float*  Hin    = (float*) alloc(sumN * 4);
    half_t* w_in_t = (half_t*)alloc((size_t)HID * 2 * D_IN * 2);
    half_t* w_out_t= (half_t*)alloc((size_t)D_IN * HID * 2);
    half_t* wx_t   = (half_t*)alloc((size_t)D_IN * DBLW * 2);
    half_t* m1t    = (half_t*)alloc((size_t)HID * MLPD * 2);
    half_t* m2t    = (half_t*)alloc((size_t)MLPD * HID * 2);
    half_t* clsin  = (half_t*)alloc((size_t)32 * HID * 2);
    half_t* h1c    = (half_t*)alloc((size_t)32 * MLPD * 2);
    float*  h2c    = (float*) alloc((size_t)32 * HID * 4);

    // per-launch weight transpose+cast to f16 (Bt layout: Npad x K)
    transpose_cast_kernel<<<(HID * 2 * D_IN + 255) / 256, 256, 0, stream>>>(w_in,  w_in_t,  HID,  2 * D_IN, 2 * D_IN);
    transpose_cast_kernel<<<(D_IN * HID + 255) / 256,     256, 0, stream>>>(w_out, w_out_t, D_IN, HID, HID);
    transpose_cast_kernel<<<(D_IN * DBLW + 255) / 256,    256, 0, stream>>>(w_xproj, wx_t,  D_IN, DTR + 2 * NST, DBLW);
    transpose_cast_kernel<<<(HID * MLPD + 255) / 256,     256, 0, stream>>>(mlp_w1, m1t,    HID,  MLPD, MLPD);
    transpose_cast_kernel<<<(MLPD * HID + 255) / 256,     256, 0, stream>>>(mlp_w2, m2t,    MLPD, HID, HID);

    // patch embedding + cls token
    patch_embed_kernel<<<dim3(B_ * LP, (HID + 255) / 256), 256, 0, stream>>>(x, patch_w, patch_b, pos_emb, tok);
    cls_init_kernel<<<dim3(B_, (HID + 255) / 256), 256, 0, stream>>>(cls_tok, tok);

    const dim3 gemmBlk(256);
    const int gemmGx = (MPAIRS + 7) / 8;
    for (int layer = 0; layer < LAYERS; ++layer) {
        rmsnorm_cast_kernel<<<MROWS, 256, 0, stream>>>(tok, norm_w, h16);
        // uz = h @ w_in   (M x 576) * (576 x 2304)
        wmma_gemm_kernel<<<dim3(gemmGx, (2 * D_IN) / 64), gemmBlk, 0, stream>>>(
            h16, w_in_t, nullptr, uz, nullptr, HID, 2 * D_IN, MPAIRS, 0);
        conv_silu_kernel<<<(B_ * LL * D_IN + 255) / 256, 256, 0, stream>>>(uz, conv_w, conv_b, u, u16);
        // dbl = u @ w_xproj (N padded 36 -> 64)
        wmma_gemm_kernel<<<dim3(gemmGx, DBLW / 64), gemmBlk, 0, stream>>>(
            u16, wx_t, nullptr, dbl, nullptr, D_IN, DBLW, MPAIRS, 0);
        dt_kernel<<<(MROWS * D_IN + 255) / 256, 256, 0, stream>>>(dbl, w_dt, b_dt, dtb);
        // chunked parallel selective scan
        scan_chunk_kernel<<<dim3(D_IN / 128, NCH, B_), 128, 0, stream>>>(u, dtb, dbl, A_log, Pbuf, Qbuf);
        scan_combine_kernel<<<(B_ * D_IN * NST + 255) / 256, 256, 0, stream>>>(Pbuf, Qbuf, Hin);
        scan_final_kernel<<<dim3(D_IN / 128, NCH, B_), 128, 0, stream>>>(u, dtb, dbl, uz, A_log, D_param, Hin, y16);
        // tok += y @ w_out   (M x 1152) * (1152 x 576), residual fused
        wmma_gemm_kernel<<<dim3(gemmGx, HID / 64), gemmBlk, 0, stream>>>(
            y16, w_out_t, tok, tok, nullptr, D_IN, HID, MPAIRS, 1);
    }

    // MLP head on token 0 of each batch only (single 32-row block)
    gather_cls_kernel<<<(32 * HID + 255) / 256, 256, 0, stream>>>(tok, clsin);
    wmma_gemm_kernel<<<dim3(1, MLPD / 64), gemmBlk, 0, stream>>>(
        clsin, m1t, mlp_b1, nullptr, h1c, HID, MLPD, 1, 2);     // gelu(x@w1+b1) -> f16
    wmma_gemm_kernel<<<dim3(1, HID / 64), gemmBlk, 0, stream>>>(
        h1c, m2t, mlp_b2, h2c, nullptr, MLPD, HID, 1, 3);       // @w2+b2 -> f32
    head_kernel<<<(B_ * NCLS + 255) / 256, 256, 0, stream>>>(h2c, cls_w, cls_b, (float*)d_out);
}
